// minLSTM_49813030699440
// MI455X (gfx1250) — compile-verified, matrix operand order checked by reading, not run on hardware
//
#include <hip/hip_runtime.h>
#include <cstddef>

#define B_    4
#define S_    8192
#define D_    512
#define BM    128
#define BN    64
#define BK    32
#define LDA   40        // padded LDS stride in bf16 elems (80B: 16B-aligned, conflict-light)
#define CHUNK 128
#define NCH   (S_ / CHUNK)   // 64 chunks
#define NCHAN (B_ * D_)      // 2048 channels

typedef __attribute__((ext_vector_type(8)))  float  v8f;
typedef __attribute__((ext_vector_type(8)))  __bf16 v8bf;
typedef __attribute__((ext_vector_type(16))) __bf16 v16bf;

__device__ __forceinline__ float log_sigmoidf(float z) {
    // log(sigmoid(z)) = -softplus(-z), numerically stable
    return fminf(z, 0.0f) - log1pf(__expf(-fabsf(z)));
}
__device__ __forceinline__ float log_gf(float z) {
    // x>=0: log(relu(x)+0.5); x<0: -softplus(-x)
    return (z >= 0.0f) ? __logf(z + 0.5f) : log_sigmoidf(z);
}
__device__ __forceinline__ float lse2(float p, float q) {
    float m = fmaxf(p, q);
    return m + log1pf(__expf(-fabsf(p - q)));
}

union FragBF {
    v16bf v;
    v8bf  h[2];
};

union PackBF2 {
    __bf16   h[2];
    unsigned u;
};

// Register staging for one (As, Bs x3) tile: 10 global b128 loads issued in a
// batch so their latency overlaps the previous tile's WMMAs.
struct StageRegs {
    float4 a[4];       // A tile: 4 float4 per thread (128x32 / 256 threads)
    float4 b0[3];      // B tile per gate: k-row 2*kp,   cols nc..nc+3
    float4 b1[3];      // B tile per gate: k-row 2*kp+1, cols nc..nc+3
};

__device__ __forceinline__ void stage_load(const float* __restrict__ x,
                                           const float* const W[3],
                                           int mBase, int n0, int kb, int tid,
                                           StageRegs& r)
{
    #pragma unroll
    for (int it = 0; it < 4; ++it) {
        int q   = it * 256 + tid;          // 1024 float4 chunks
        int row = q >> 3;
        int kq  = (q & 7) << 2;
        r.a[it] = *reinterpret_cast<const float4*>(
            x + (size_t)(mBase + row) * D_ + kb + kq);
    }
    const int kp = tid >> 4;               // 0..15: k-pair index
    const int nc = (tid & 15) << 2;        // 0..60: column base
    #pragma unroll
    for (int g = 0; g < 3; ++g) {
        const float* wp = W[g] + (size_t)(kb + 2 * kp) * D_ + n0 + nc;
        r.b0[g] = *reinterpret_cast<const float4*>(wp);
        r.b1[g] = *reinterpret_cast<const float4*>(wp + D_);
    }
}

__device__ __forceinline__ void stage_store(__bf16* __restrict__ As,
                                            __bf16* __restrict__ Bs, // [3][BN*LDA]
                                            int tid, const StageRegs& r)
{
    #pragma unroll
    for (int it = 0; it < 4; ++it) {
        int q   = it * 256 + tid;
        int row = q >> 3;
        int kq  = (q & 7) << 2;
        __bf16* dst = &As[row * LDA + kq];
        dst[0] = (__bf16)r.a[it].x; dst[1] = (__bf16)r.a[it].y;
        dst[2] = (__bf16)r.a[it].z; dst[3] = (__bf16)r.a[it].w;
    }
    const int kp = tid >> 4;
    const int nc = (tid & 15) << 2;
    #pragma unroll
    for (int g = 0; g < 3; ++g) {
        const float* lo = reinterpret_cast<const float*>(&r.b0[g]);
        const float* hi = reinterpret_cast<const float*>(&r.b1[g]);
        #pragma unroll
        for (int j = 0; j < 4; ++j) {
            PackBF2 pk;
            pk.h[0] = (__bf16)lo[j];       // k = 2*kp
            pk.h[1] = (__bf16)hi[j];       // k = 2*kp+1  (k-adjacent in transposed layout)
            *reinterpret_cast<unsigned*>(
                &Bs[g * (BN * LDA) + (nc + j) * LDA + 2 * kp]) = pk.u;
        }
    }
}

// ---------------------------------------------------------------------------
// Kernel 1: fused 3-way GEMM (bf16 WMMA, fp32 accum) + log-space activations.
//   outA (= d_out scratch) <- log_f
//   outV (= workspace)     <- log_i + log_g(h~)
// Double-buffered LDS, register-staged global loads.
// ---------------------------------------------------------------------------
__global__ __launch_bounds__(256)
void minlstm_gates_gemm(const float* __restrict__ x,
                        const float* __restrict__ Wf, const float* __restrict__ bfp,
                        const float* __restrict__ Wi, const float* __restrict__ bip,
                        const float* __restrict__ Wh, const float* __restrict__ bhp,
                        float* __restrict__ outA,
                        float* __restrict__ outV)
{
    __shared__ __attribute__((aligned(16))) __bf16 As[2][BM * LDA];
    __shared__ __attribute__((aligned(16))) __bf16 Bs[2][3][BN * LDA];

    const int tid   = threadIdx.x;
    const int lane  = tid & 31;
    const int wave  = tid >> 5;      // 8 waves
    const int wm    = wave & 3;      // 4 waves over M
    const int wn    = wave >> 2;     // 2 waves over N
    const int mBase = blockIdx.x * BM;
    const int n0    = blockIdx.y * BN;

    const float* W[3] = {Wf, Wi, Wh};

    v8f acc[3][2][2];
    #pragma unroll
    for (int g = 0; g < 3; ++g)
        #pragma unroll
        for (int i = 0; i < 2; ++i)
            #pragma unroll
            for (int j = 0; j < 2; ++j)
                acc[g][i][j] = v8f{};

    const int lrow = lane & 15;
    const int kb2  = (lane >> 4) << 3;   // 0 for lanes 0-15, 8 for lanes 16-31

    StageRegs r;
    stage_load(x, W, mBase, n0, 0, tid, r);

    int buf = 0;
    for (int kb = 0; kb < D_; kb += BK) {
        stage_store(As[buf], &Bs[buf][0][0], tid, r);
        __syncthreads();

        // issue next tile's global loads; latency hidden under WMMAs below
        if (kb + BK < D_)
            stage_load(x, W, mBase, n0, kb + BK, tid, r);

        FragBF aF[2];
        #pragma unroll
        for (int i = 0; i < 2; ++i) {
            int rr = wm * 32 + i * 16 + lrow;
            aF[i].h[0] = *reinterpret_cast<const v8bf*>(&As[buf][rr * LDA + kb2]);
            aF[i].h[1] = *reinterpret_cast<const v8bf*>(&As[buf][rr * LDA + 16 + kb2]);
        }
        #pragma unroll
        for (int g = 0; g < 3; ++g) {
            #pragma unroll
            for (int j = 0; j < 2; ++j) {
                FragBF bF;
                int c = wn * 32 + j * 16 + lrow;
                bF.h[0] = *reinterpret_cast<const v8bf*>(&Bs[buf][g][c * LDA + kb2]);
                bF.h[1] = *reinterpret_cast<const v8bf*>(&Bs[buf][g][c * LDA + 16 + kb2]);
                #pragma unroll
                for (int i = 0; i < 2; ++i) {
                    acc[g][i][j] = __builtin_amdgcn_wmma_f32_16x16x32_bf16(
                        false, aF[i].v, false, bF.v, (short)0, acc[g][i][j],
                        false, false);
                }
            }
        }
        buf ^= 1;
        // Second barrier is redundant: the barrier at the top of the next
        // iteration (after stage_store) orders compute(kb) before store(kb+2)
        // into the same buffer.
        __syncthreads();
    }

    // ---- epilogue: bias + activations ----
    const int rhi = (lane >> 4) << 3;    // C/D layout: hi lanes hold rows +8
    #pragma unroll
    for (int j = 0; j < 2; ++j) {
        int col = n0 + wn * 32 + j * 16 + lrow;
        float bfv = bfp[col], biv = bip[col], bhv = bhp[col];
        #pragma unroll
        for (int i = 0; i < 2; ++i) {
            int rowb = mBase + wm * 32 + i * 16 + rhi;
            #pragma unroll
            for (int rr = 0; rr < 8; ++rr) {
                size_t idx = (size_t)(rowb + rr) * D_ + col;
                float yf = acc[0][i][j][rr] + bfv;
                float yi = acc[1][i][j][rr] + biv;
                float yh = acc[2][i][j][rr] + bhv;
                outA[idx] = log_sigmoidf(yf);
                outV[idx] = log_sigmoidf(yi) + log_gf(yh);
            }
        }
    }
}

// ---------------------------------------------------------------------------
// Kernel 2: per-(channel, chunk) reduction of the scan operator
//   (A,V) ∘ (a,v) = (A+a, lse(V+a, v))
// ---------------------------------------------------------------------------
__global__ __launch_bounds__(256)
void minlstm_chunk_reduce(const float* __restrict__ a,   // d_out scratch (log_f)
                          const float* __restrict__ v,   // workspace
                          float* __restrict__ sumA,
                          float* __restrict__ sumV)
{
    const int d = blockIdx.x * 256 + threadIdx.x;
    const int c = blockIdx.y;
    const int b = blockIdx.z;
    const size_t base = ((size_t)b * S_ + (size_t)c * CHUNK) * D_ + d;

    float A = a[base];
    float V = v[base];
    for (int t = 1; t < CHUNK; ++t) {
        size_t idx = base + (size_t)t * D_;
        __builtin_prefetch(a + idx + 16 * (size_t)D_, 0, 1);
        __builtin_prefetch(v + idx + 16 * (size_t)D_, 0, 1);
        float at = a[idx];
        float vt = v[idx];
        A += at;
        V  = lse2(V + at, vt);
    }
    const int su = c * NCHAN + b * D_ + d;
    sumA[su] = A;
    sumV[su] = V;
}

// ---------------------------------------------------------------------------
// Kernel 3: exclusive scan over chunk aggregates, seeded with log_g(pre_h)
// ---------------------------------------------------------------------------
__global__ __launch_bounds__(256)
void minlstm_chunk_prefix(const float* __restrict__ pre_h,
                          const float* __restrict__ sumA,
                          const float* __restrict__ sumV,
                          float* __restrict__ pref)
{
    const int ch = blockIdx.x * 256 + threadIdx.x;   // 0..NCHAN-1
    float s = log_gf(pre_h[ch]);                     // initial log-state
    for (int c = 0; c < NCH; ++c) {
        pref[c * NCHAN + ch] = s;                    // state at chunk start
        s = lse2(sumA[c * NCHAN + ch] + s, sumV[c * NCHAN + ch]);
    }
}

// ---------------------------------------------------------------------------
// Kernel 4: apply pass — recompute scan inside each chunk from its prefix,
// overwrite d_out (which held log_f) with h = exp(s).
// ---------------------------------------------------------------------------
__global__ __launch_bounds__(256)
void minlstm_scan_apply(float* __restrict__ aOut,        // in: log_f, out: h
                        const float* __restrict__ v,
                        const float* __restrict__ pref)
{
    const int d = blockIdx.x * 256 + threadIdx.x;
    const int c = blockIdx.y;
    const int b = blockIdx.z;
    const size_t base = ((size_t)b * S_ + (size_t)c * CHUNK) * D_ + d;

    float s = pref[c * NCHAN + b * D_ + d];
    for (int t = 0; t < CHUNK; ++t) {
        size_t idx = base + (size_t)t * D_;
        __builtin_prefetch(aOut + idx + 16 * (size_t)D_, 0, 1);
        __builtin_prefetch(v + idx + 16 * (size_t)D_, 0, 1);
        float at = aOut[idx];
        float vt = v[idx];
        s = lse2(at + s, vt);
        aOut[idx] = expf(s);
    }
}

extern "C" void kernel_launch(void* const* d_in, const int* in_sizes, int n_in,
                              void* d_out, int out_size, void* d_ws, size_t ws_size,
                              hipStream_t stream) {
    (void)in_sizes; (void)n_in; (void)out_size; (void)ws_size;
    const float* x     = (const float*)d_in[0];
    const float* pre_h = (const float*)d_in[1];
    const float* Wf    = (const float*)d_in[2];
    const float* bfp   = (const float*)d_in[3];
    const float* Wi    = (const float*)d_in[4];
    const float* bip   = (const float*)d_in[5];
    const float* Wh    = (const float*)d_in[6];
    const float* bhp   = (const float*)d_in[7];

    float* out  = (float*)d_out;                       // scratch log_f, final h
    float* wsV  = (float*)d_ws;                        // B*S*D floats (64 MB)
    float* sumA = wsV  + (size_t)B_ * S_ * D_;         // NCH*NCHAN floats
    float* sumV = sumA + (size_t)NCH * NCHAN;
    float* pref = sumV + (size_t)NCH * NCHAN;

    // 1) fused tri-GEMM + activations
    minlstm_gates_gemm<<<dim3((B_ * S_) / BM, D_ / BN), 256, 0, stream>>>(
        x, Wf, bfp, Wi, bip, Wh, bhp, out, wsV);

    // 2) chunk-local reductions
    minlstm_chunk_reduce<<<dim3(D_ / 256, NCH, B_), 256, 0, stream>>>(
        out, wsV, sumA, sumV);

    // 3) scan over chunk aggregates
    minlstm_chunk_prefix<<<dim3(NCHAN / 256), 256, 0, stream>>>(
        pre_h, sumA, sumV, pref);

    // 4) apply: emit h = exp(scan state)
    minlstm_scan_apply<<<dim3(D_ / 256, NCH, B_), 256, 0, stream>>>(
        out, wsV, pref);
}
